// MultiHeadCrossAttn_31576599560657
// MI455X (gfx1250) — matrix operand order repaired and checked by reading, not compiled, not run on hardware
//
#include <hip/hip_runtime.h>
#include <hip/hip_bf16.h>
#include <stdint.h>
#include <stddef.h>

// Problem constants (match reference)
#define B_   256
#define N_   128
#define D_   200
#define DG_  256
#define DT_  256
#define DK_  256
#define H_   4
#define DH_  64
#define EPS_ 1e-5f
#define TPAD 208   // max token count padded to multiple of 16 (200 -> 208)

typedef __attribute__((ext_vector_type(16))) __bf16 v16bf;
typedef __attribute__((ext_vector_type(8)))  float  v8f;

// ---------------------------------------------------------------------------
// Optional CDNA5 async global->LDS copy path (GLOBAL_LOAD_ASYNC_TO_LDS_B128,
// tracked by ASYNCcnt). Guarded so the file compiles on toolchains that do
// not declare the builtin; fallback is VGPR staging.
// Builtin signature (from clang diagnostic): (v4i AS1*, v4i AS3*, Ii, Ii)
// ---------------------------------------------------------------------------
#if defined(__has_builtin)
#  if __has_builtin(__builtin_amdgcn_global_load_async_to_lds_b128)
#    define HAVE_ASYNC_LDS 1
#  endif
#endif
#ifndef HAVE_ASYNC_LDS
#  define HAVE_ASYNC_LDS 0
#endif

#if HAVE_ASYNC_LDS
typedef int v4i_async __attribute__((vector_size(16)));
typedef __attribute__((address_space(1))) v4i_async* g_v4i_ptr;
typedef __attribute__((address_space(3))) v4i_async* l_v4i_ptr;
__device__ __forceinline__ void async_ld_b128(const void* g, void* l) {
  __builtin_amdgcn_global_load_async_to_lds_b128(
      (g_v4i_ptr)(uintptr_t)g, (l_v4i_ptr)(uint32_t)(uintptr_t)l, 0, 0);
}
__device__ __forceinline__ void async_wait0() {
  asm volatile("s_wait_asynccnt 0x0" ::: "memory");
}
#endif

// ---------------------------------------------------------------------------
// WMMA fragment helpers (operand layouts this code uses for v_wmma_f32_16x16x32_bf16, wave32)
// A (16x32, MxK): lane L holds row L%16; elems i -> K = (L/16)*16 + i.
// B (32x16, KxN): lane L holds col L%16; elems i -> K = (L/16)*16 + i.
// C (16x16 f32):  elem r -> row r + (lane>=16 ? 8 : 0), col = lane%16.
// ---------------------------------------------------------------------------
__device__ __forceinline__ v16bf wmma_load_a(const __bf16* base, int ld, int lane) {
  int row  = lane & 15;
  int koff = (lane >> 4) << 4;
  const __bf16* p0 = base + row * ld + koff;
  v16bf a;
#pragma unroll
  for (int i = 0; i < 16; ++i) a[i] = p0[i];
  return a;
}

// B staged transposed in LDS: baseT[n][k], row stride ld (k-contiguous)
__device__ __forceinline__ v16bf wmma_load_bT(const __bf16* baseT, int ld, int lane) {
  int n    = lane & 15;
  int koff = (lane >> 4) << 4;
  const __bf16* p = baseT + n * ld + koff;
  v16bf b;
#pragma unroll
  for (int i = 0; i < 16; ++i) b[i] = p[i];
  return b;
}

__device__ __forceinline__ v8f wmma_bf16(v16bf a, v16bf b, v8f c) {
  return __builtin_amdgcn_wmma_f32_16x16x32_bf16(false, a, false, b, (short)0, c,
                                                 false, false);
}

// Block-wide LayerNorm over 256 channels (blockDim.x == 256). red: float[512].
__device__ __forceinline__ float block_ln_256(float x, float* red,
                                              const float* g, const float* b, int t) {
  red[t] = x; red[256 + t] = x * x;
  __syncthreads();
  for (int o = 128; o > 0; o >>= 1) {
    if (t < o) { red[t] += red[t + o]; red[256 + t] += red[256 + t + o]; }
    __syncthreads();
  }
  float mu  = red[0]   * (1.0f / 256.0f);
  float var = red[256] * (1.0f / 256.0f) - mu * mu;
  __syncthreads();
  return (x - mu) * rsqrtf(var + EPS_) * g[t] + b[t];
}

// ---------------------------------------------------------------------------
// f32 -> bf16 conversion
// ---------------------------------------------------------------------------
__global__ void cvt_f32_bf16_kernel(const float* __restrict__ in,
                                    __bf16* __restrict__ out, int n) {
  int i = blockIdx.x * blockDim.x + threadIdx.x;
  if (i < n) out[i] = (__bf16)in[i];
}

// ---------------------------------------------------------------------------
// Descriptor tokenizer: per (b,d) token. relu(desc*w1+b1)@w2+b2+id_emb, LN, LN.
// grid = B*D, block = 256
// ---------------------------------------------------------------------------
__global__ void tokenizer_kernel(const float* __restrict__ desc,
                                 const float* __restrict__ id_emb,
                                 const float* __restrict__ w1, const float* __restrict__ b1,
                                 const float* __restrict__ w2, const float* __restrict__ b2,
                                 const float* __restrict__ tg, const float* __restrict__ tb,
                                 const float* __restrict__ dg, const float* __restrict__ db,
                                 __bf16* __restrict__ desc_tks_bf,
                                 __bf16* __restrict__ desc_in_bf) {
  __shared__ float hsh[128];
  __shared__ float red[512];
  int token = blockIdx.x;
  int d = token % D_;
  int t = threadIdx.x;
  float x = desc[token];
  if (t < 128) { float v = fmaf(x, w1[t], b1[t]); hsh[t] = v > 0.f ? v : 0.f; }
  __syncthreads();
  float acc = b2[t];
#pragma unroll 4
  for (int k = 0; k < 128; ++k) acc = fmaf(hsh[k], w2[k * 256 + t], acc);
  acc += id_emb[d * 256 + t];
  __syncthreads();
  float tk = block_ln_256(acc, red, tg, tb, t);
  desc_tks_bf[(size_t)token * 256 + t] = (__bf16)tk;
  float di = block_ln_256(tk, red, dg, db, t);
  desc_in_bf[(size_t)token * 256 + t] = (__bf16)di;
}

// ---------------------------------------------------------------------------
// Node LN: store raw node_tokens as bf16 and LN(node_tokens) as bf16.
// grid = B*N, block = 256
// ---------------------------------------------------------------------------
__global__ void node_ln_kernel(const float* __restrict__ node_tokens,
                               const float* __restrict__ g, const float* __restrict__ b,
                               __bf16* __restrict__ node_bf,
                               __bf16* __restrict__ node_in_bf) {
  __shared__ float red[512];
  int token = blockIdx.x;
  int t = threadIdx.x;
  float x = node_tokens[(size_t)token * 256 + t];
  node_bf[(size_t)token * 256 + t] = (__bf16)x;
  float y = block_ln_256(x, red, g, b, t);
  node_in_bf[(size_t)token * 256 + t] = (__bf16)y;
}

// ---------------------------------------------------------------------------
// WMMA GEMM: C[M x 256] = A[M x 256] (bf16) @ W[256 x 256] (bf16) + bias
// Block tile 128x64, 8 waves of 32x32 (2x2 wmma). grid = (M/128, 4), block 256.
// M is always a multiple of 128 -> no row bounds checks anywhere.
// ---------------------------------------------------------------------------
__global__ void gemm256_kernel(const __bf16* __restrict__ A,
                               const __bf16* __restrict__ W,
                               const float* __restrict__ bias,
                               float* __restrict__ Cf, __bf16* __restrict__ Cb,
                               int M, int write_bf) {
  __shared__ __bf16 As[128 * 32];  // row-major [m][k]
  __shared__ __bf16 Bs[64 * 32];   // transposed [n][k]
  int m0 = blockIdx.x * 128;
  int n0 = blockIdx.y * 64;
  int t = threadIdx.x;
  int wid = t >> 5, lane = t & 31;
  int wm = (wid & 3) * 32;
  int wn = (wid >> 2) * 32;
  v8f a00 = {}, a01 = {}, a10 = {}, a11 = {};

  for (int k0 = 0; k0 < 256; k0 += 32) {
#if HAVE_ASYNC_LDS
    // A tile: 128 rows x 64B = 512 x 16B chunks, async DMA global->LDS
#pragma unroll
    for (int chunk = t; chunk < 512; chunk += 256) {
      int r = chunk >> 2;
      int coff = (chunk & 3) << 3;  // 0,8,16,24 (bf16 elems) = 16B steps
      async_ld_b128(A + (size_t)(m0 + r) * 256 + k0 + coff, &As[r * 32 + coff]);
    }
#else
    for (int idx = t; idx < 128 * 32; idx += 256) {
      int r = idx >> 5, c = idx & 31;
      As[idx] = A[(size_t)(m0 + r) * 256 + k0 + c];
    }
#endif
    // B tile needs a transpose -> VGPR staging
    for (int idx = t; idx < 64 * 32; idx += 256) {
      int k = idx >> 6, n = idx & 63;
      Bs[n * 32 + k] = W[(size_t)(k0 + k) * 256 + n0 + n];
    }
#if HAVE_ASYNC_LDS
    async_wait0();
#endif
    __syncthreads();
    v16bf fa0 = wmma_load_a(As + wm * 32, 32, lane);
    v16bf fa1 = wmma_load_a(As + (wm + 16) * 32, 32, lane);
    v16bf fb0 = wmma_load_bT(Bs + wn * 32, 32, lane);
    v16bf fb1 = wmma_load_bT(Bs + (wn + 16) * 32, 32, lane);
    a00 = wmma_bf16(fa0, fb0, a00);
    a01 = wmma_bf16(fa0, fb1, a01);
    a10 = wmma_bf16(fa1, fb0, a10);
    a11 = wmma_bf16(fa1, fb1, a11);
    __syncthreads();
  }

  int lrow = (lane >> 4) * 8;
  int lcol = lane & 15;
  auto store_tile = [&](v8f c, int ms, int ns) {
    int colg = n0 + wn + ns + lcol;
    float bv = bias[colg];
#pragma unroll
    for (int r = 0; r < 8; ++r) {
      int rowg = m0 + wm + ms + lrow + r;
      float v = c[r] + bv;
      if (write_bf) Cb[(size_t)rowg * 256 + colg] = (__bf16)v;
      else          Cf[(size_t)rowg * 256 + colg] = v;
    }
  };
  store_tile(a00, 0, 0);
  store_tile(a01, 0, 16);
  store_tile(a10, 16, 0);
  store_tile(a11, 16, 16);
  (void)M;
}

// ---------------------------------------------------------------------------
// Attention scores: per (b,h) block. scores = (Q @ K^T) / 8, masked keys -> -inf
// Q rows: b*Tq..(+Tq), stride 256, head slice h*64. grid = B*H, block = 256.
// ---------------------------------------------------------------------------
__global__ void attn_scores_kernel(const __bf16* __restrict__ Qall,
                                   const __bf16* __restrict__ Kall,
                                   const int* __restrict__ keymask,
                                   float* __restrict__ scores, int Tq, int Tk) {
  __shared__ __bf16 Qs[TPAD * 64];
  __shared__ __bf16 Ks[TPAD * 64];
  int bh = blockIdx.x;
  int b = bh >> 2, h = bh & 3;
  int t = threadIdx.x;
  int wid = t >> 5, lane = t & 31;
  const __bf16* Qb = Qall + (size_t)b * Tq * 256 + h * 64;
  const __bf16* Kb = Kall + (size_t)b * Tk * 256 + h * 64;

#if HAVE_ASYNC_LDS
  // rows are 128B (8 x 16B chunks); real rows via async DMA, pad rows zeroed
  for (int chunk = t; chunk < TPAD * 8; chunk += 256) {
    int r = chunk >> 3;
    int coff = (chunk & 7) << 3;  // bf16 elems, 16B steps
    if (r < Tq) {
      async_ld_b128(Qb + (size_t)r * 256 + coff, &Qs[r * 64 + coff]);
    } else {
      __bf16* z = &Qs[r * 64 + coff];
#pragma unroll
      for (int i = 0; i < 8; ++i) z[i] = (__bf16)0.f;
    }
    if (r < Tk) {
      async_ld_b128(Kb + (size_t)r * 256 + coff, &Ks[r * 64 + coff]);
    } else {
      __bf16* z = &Ks[r * 64 + coff];
#pragma unroll
      for (int i = 0; i < 8; ++i) z[i] = (__bf16)0.f;
    }
  }
  async_wait0();
#else
  for (int idx = t; idx < TPAD * 64; idx += 256) {
    int r = idx >> 6, c = idx & 63;
    Qs[idx] = (r < Tq) ? Qb[(size_t)r * 256 + c] : (__bf16)0.f;
    Ks[idx] = (r < Tk) ? Kb[(size_t)r * 256 + c] : (__bf16)0.f;
  }
#endif
  __syncthreads();

  int nQt = (Tq + 15) >> 4;
  int nKt = (Tk + 15) >> 4;
  int lrow = (lane >> 4) * 8;
  int lcol = lane & 15;

  for (int qt = wid; qt < nQt; qt += 8) {
    v16bf aq0 = wmma_load_a(Qs + qt * 16 * 64, 64, lane);
    v16bf aq1 = wmma_load_a(Qs + qt * 16 * 64 + 32, 64, lane);
    for (int kt = 0; kt < nKt; ++kt) {
      // K rows are token-major with contiguous head dims == transposed-B layout
      v16bf bk0 = wmma_load_bT(Ks + kt * 16 * 64, 64, lane);
      v16bf bk1 = wmma_load_bT(Ks + kt * 16 * 64 + 32, 64, lane);
      v8f c = {};
      c = wmma_bf16(aq0, bk0, c);
      c = wmma_bf16(aq1, bk1, c);
      int col = kt * 16 + lcol;
      if (col < Tk) {
        bool masked = (keymask != nullptr) && (keymask[b * Tk + col] != 0);
#pragma unroll
        for (int r = 0; r < 8; ++r) {
          int q = qt * 16 + lrow + r;
          if (q < Tq) {
            float v = masked ? -__builtin_inff() : c[r] * 0.125f;  // 1/sqrt(64)
            scores[((size_t)bh * Tq + q) * Tk + col] = v;
          }
        }
      }
    }
  }
}

// ---------------------------------------------------------------------------
// Row softmax: grid = B*H*Tq rows, block = 256 (Tk <= 256)
// ---------------------------------------------------------------------------
__global__ void softmax_kernel(const float* __restrict__ scores,
                               float* __restrict__ attn, int Tk) {
  __shared__ float red[256];
  size_t row = blockIdx.x;
  int t = threadIdx.x;
  float v = (t < Tk) ? scores[row * Tk + t] : -__builtin_inff();
  red[t] = v;
  __syncthreads();
  for (int o = 128; o > 0; o >>= 1) {
    if (t < o) red[t] = fmaxf(red[t], red[t + o]);
    __syncthreads();
  }
  float m = red[0];
  __syncthreads();
  float e = (t < Tk) ? __expf(v - m) : 0.f;
  red[t] = e;
  __syncthreads();
  for (int o = 128; o > 0; o >>= 1) {
    if (t < o) red[t] += red[t + o];
    __syncthreads();
  }
  float s = red[0];
  if (t < Tk) attn[row * Tk + t] = e / s;
}

// ---------------------------------------------------------------------------
// Context: per (b,h). ctx = attn (Tq x Tk, f32->bf16) @ V (Tk x 64, bf16).
// K-loop chunks of 32 keys through LDS. Writes ctx[b][q][h*64 + n] (f32).
// grid = B*H, block = 256.
// ---------------------------------------------------------------------------
__global__ void attn_ctx_kernel(const float* __restrict__ attn,
                                const __bf16* __restrict__ Vall,
                                float* __restrict__ ctx, int Tq, int Tk) {
  __shared__ __bf16 As[TPAD * 32];  // [q][kk]
  __shared__ __bf16 Vts[64 * 32];   // transposed [n][kk]
  int bh = blockIdx.x;
  int b = bh >> 2, h = bh & 3;
  int t = threadIdx.x;
  int wid = t >> 5, lane = t & 31;
  const float* Ab = attn + (size_t)bh * Tq * Tk;
  const __bf16* Vb = Vall + (size_t)b * Tk * 256 + h * 64;
  int nQt = (Tq + 15) >> 4;
  v8f acc[2][4] = {};

  for (int k0 = 0; k0 < Tk; k0 += 32) {
    // attn chunk needs f32->bf16 conversion; V chunk needs transpose: VGPR path
    for (int idx = t; idx < TPAD * 32; idx += 256) {
      int q = idx >> 5, kk = idx & 31;
      float v = (q < Tq && (k0 + kk) < Tk) ? Ab[(size_t)q * Tk + k0 + kk] : 0.f;
      As[idx] = (__bf16)v;
    }
    for (int idx = t; idx < 64 * 32; idx += 256) {
      int kk = idx >> 6, n = idx & 63;
      Vts[n * 32 + kk] = ((k0 + kk) < Tk) ? Vb[(size_t)(k0 + kk) * 256 + n]
                                          : (__bf16)0.f;
    }
    __syncthreads();
#pragma unroll
    for (int s = 0; s < 2; ++s) {
      int qt = wid + s * 8;
      if (qt < nQt) {
        v16bf fa = wmma_load_a(As + qt * 16 * 32, 32, lane);
#pragma unroll
        for (int nt = 0; nt < 4; ++nt) {
          v16bf fb = wmma_load_bT(Vts + nt * 16 * 32, 32, lane);
          acc[s][nt] = wmma_bf16(fa, fb, acc[s][nt]);
        }
      }
    }
    __syncthreads();
  }

  int lrow = (lane >> 4) * 8;
  int lcol = lane & 15;
#pragma unroll
  for (int s = 0; s < 2; ++s) {
    int qt = wid + s * 8;
    if (qt < nQt) {
#pragma unroll
      for (int nt = 0; nt < 4; ++nt) {
#pragma unroll
        for (int r = 0; r < 8; ++r) {
          int q = qt * 16 + lrow + r;
          if (q < Tq)
            ctx[((size_t)b * Tq + q) * 256 + h * 64 + nt * 16 + lcol] = acc[s][nt][r];
        }
      }
    }
  }
}

// ---------------------------------------------------------------------------
// out_tokens = LN(P + ctx).  grid = B*T, block = 256
// ---------------------------------------------------------------------------
__global__ void out_tokens_kernel(const float* __restrict__ P,
                                  const float* __restrict__ ctx,
                                  const float* __restrict__ g,
                                  const float* __restrict__ b,
                                  float* __restrict__ out) {
  __shared__ float red[512];
  size_t token = blockIdx.x;
  int t = threadIdx.x;
  float x = P[token * 256 + t] + ctx[token * 256 + t];
  out[token * 256 + t] = block_ln_256(x, red, g, b, t);
}

// g_vec: masked mean over N nodes. grid = B, block = 256
__global__ void gvec_kernel(const float* __restrict__ outg,
                            const int* __restrict__ mask,
                            float* __restrict__ gvec) {
  int b = blockIdx.x, t = threadIdx.x;
  float s = 0.f;
  int cnt = 0;
  for (int n = 0; n < N_; ++n) {
    if (mask[b * N_ + n] == 0) {
      s += outg[((size_t)b * N_ + n) * 256 + t];
      cnt++;
    }
  }
  float denom = (float)(cnt > 0 ? cnt : 1);
  gvec[b * 256 + t] = s / denom;
}

// d_vec: mean over D descriptors. grid = B, block = 256
__global__ void dvec_kernel(const float* __restrict__ outd,
                            float* __restrict__ dvec) {
  int b = blockIdx.x, t = threadIdx.x;
  float s = 0.f;
  for (int d = 0; d < D_; ++d) s += outd[((size_t)b * D_ + d) * 256 + t];
  dvec[b * 256 + t] = s * (1.0f / (float)D_);
}

// ---------------------------------------------------------------------------
// Launch
// ---------------------------------------------------------------------------
extern "C" void kernel_launch(void* const* d_in, const int* in_sizes, int n_in,
                              void* d_out, int out_size, void* d_ws, size_t ws_size,
                              hipStream_t stream) {
  const float* node_tokens = (const float*)d_in[0];
  const float* desc        = (const float*)d_in[1];
  const int*   pad_mask    = (const int*)d_in[2];
  const float* id_emb      = (const float*)d_in[3];
  const float* val_w1      = (const float*)d_in[4];
  const float* val_b1      = (const float*)d_in[5];
  const float* val_w2      = (const float*)d_in[6];
  const float* val_b2      = (const float*)d_in[7];
  const float* tok_ln_g    = (const float*)d_in[8];
  const float* tok_ln_b    = (const float*)d_in[9];
  const float* ln_node_g   = (const float*)d_in[10];
  const float* ln_node_b   = (const float*)d_in[11];
  const float* ln_desc_g   = (const float*)d_in[12];
  const float* ln_desc_b   = (const float*)d_in[13];
  const float* Wq_g = (const float*)d_in[14]; const float* bq_g = (const float*)d_in[15];
  const float* Wk_g = (const float*)d_in[16]; const float* bk_g = (const float*)d_in[17];
  const float* Wv_g = (const float*)d_in[18]; const float* bv_g = (const float*)d_in[19];
  const float* Wq_d = (const float*)d_in[20]; const float* bq_d = (const float*)d_in[21];
  const float* Wk_d = (const float*)d_in[22]; const float* bk_d = (const float*)d_in[23];
  const float* Wv_d = (const float*)d_in[24]; const float* bv_d = (const float*)d_in[25];
  const float* proj_node_w = (const float*)d_in[26]; const float* proj_node_b = (const float*)d_in[27];
  const float* proj_desc_w = (const float*)d_in[28]; const float* proj_desc_b = (const float*)d_in[29];
  const float* ln_out_g_g = (const float*)d_in[30]; const float* ln_out_g_b = (const float*)d_in[31];
  const float* ln_out_d_g = (const float*)d_in[32]; const float* ln_out_d_b = (const float*)d_in[33];

  const size_t Mnode = (size_t)B_ * N_;  // 32768
  const size_t Mdesc = (size_t)B_ * D_;  // 51200

  // ---- workspace bump allocator ----
  char* ws = (char*)d_ws;
  size_t off = 0;
  auto alloc = [&](size_t bytes) -> void* {
    void* p = ws + off;
    off = (off + bytes + 255) & ~(size_t)255;
    return p;
  };
  __bf16* Wbf[8];
  for (int i = 0; i < 8; ++i) Wbf[i] = (__bf16*)alloc(65536 * sizeof(__bf16));
  __bf16* desc_tks_bf = (__bf16*)alloc(Mdesc * 256 * sizeof(__bf16));
  __bf16* desc_in_bf  = (__bf16*)alloc(Mdesc * 256 * sizeof(__bf16));
  __bf16* node_bf     = (__bf16*)alloc(Mnode * 256 * sizeof(__bf16));
  __bf16* node_in_bf  = (__bf16*)alloc(Mnode * 256 * sizeof(__bf16));
  __bf16* Qg = (__bf16*)alloc(Mnode * 256 * sizeof(__bf16));
  __bf16* Kg = (__bf16*)alloc(Mdesc * 256 * sizeof(__bf16));
  __bf16* Vg = (__bf16*)alloc(Mdesc * 256 * sizeof(__bf16));
  __bf16* Qd = (__bf16*)alloc(Mdesc * 256 * sizeof(__bf16));
  __bf16* Kd = (__bf16*)alloc(Mnode * 256 * sizeof(__bf16));
  __bf16* Vd = (__bf16*)alloc(Mnode * 256 * sizeof(__bf16));
  float* Pn   = (float*)alloc(Mnode * 256 * sizeof(float));
  float* Pd   = (float*)alloc(Mdesc * 256 * sizeof(float));
  float* ctxg = (float*)alloc(Mnode * 256 * sizeof(float));
  float* ctxd = (float*)alloc(Mdesc * 256 * sizeof(float));

  // ---- d_out layout (flat f32, reference return order) ----
  float* fout = (float*)d_out;
  size_t o = 0;
  float* gvec = fout + o; o += (size_t)B_ * DK_;
  float* dvec = fout + o; o += (size_t)B_ * DK_;
  float* outg = fout + o; o += Mnode * DK_;
  float* outd = fout + o; o += Mdesc * DK_;
  float* sg = fout + o;   o += (size_t)B_ * H_ * N_ * D_;
  float* ag = fout + o;   o += (size_t)B_ * H_ * N_ * D_;
  float* sd = fout + o;   o += (size_t)B_ * H_ * D_ * N_;
  float* ad = fout + o;   o += (size_t)B_ * H_ * D_ * N_;

  // 1) weights -> bf16
  const float* wsrc[8] = {Wq_g, Wk_g, Wv_g, Wq_d, Wk_d, Wv_d, proj_node_w, proj_desc_w};
  for (int i = 0; i < 8; ++i)
    cvt_f32_bf16_kernel<<<256, 256, 0, stream>>>(wsrc[i], Wbf[i], 65536);

  // 2) tokenizer + node LN
  tokenizer_kernel<<<(int)Mdesc, 256, 0, stream>>>(
      desc, id_emb, val_w1, val_b1, val_w2, val_b2, tok_ln_g, tok_ln_b,
      ln_desc_g, ln_desc_b, desc_tks_bf, desc_in_bf);
  node_ln_kernel<<<(int)Mnode, 256, 0, stream>>>(
      node_tokens, ln_node_g, ln_node_b, node_bf, node_in_bf);

  // 3) projection GEMMs (WMMA)
  dim3 gN((unsigned)(Mnode / 128), 4);
  dim3 gD((unsigned)(Mdesc / 128), 4);
  gemm256_kernel<<<gN, 256, 0, stream>>>(node_in_bf, Wbf[0], bq_g, nullptr, Qg, (int)Mnode, 1);
  gemm256_kernel<<<gD, 256, 0, stream>>>(desc_in_bf, Wbf[1], bk_g, nullptr, Kg, (int)Mdesc, 1);
  gemm256_kernel<<<gD, 256, 0, stream>>>(desc_in_bf, Wbf[2], bv_g, nullptr, Vg, (int)Mdesc, 1);
  gemm256_kernel<<<gD, 256, 0, stream>>>(desc_in_bf, Wbf[3], bq_d, nullptr, Qd, (int)Mdesc, 1);
  gemm256_kernel<<<gN, 256, 0, stream>>>(node_in_bf, Wbf[4], bk_d, nullptr, Kd, (int)Mnode, 1);
  gemm256_kernel<<<gN, 256, 0, stream>>>(node_in_bf, Wbf[5], bv_d, nullptr, Vd, (int)Mnode, 1);
  gemm256_kernel<<<gN, 256, 0, stream>>>(node_bf, Wbf[6], proj_node_b, Pn, nullptr, (int)Mnode, 0);
  gemm256_kernel<<<gD, 256, 0, stream>>>(desc_tks_bf, Wbf[7], proj_desc_b, Pd, nullptr, (int)Mdesc, 0);

  // 4) attention g (nodes attend to desc, no key mask)
  attn_scores_kernel<<<B_ * H_, 256, 0, stream>>>(Qg, Kg, nullptr, sg, N_, D_);
  softmax_kernel<<<B_ * H_ * N_, 256, 0, stream>>>(sg, ag, D_);
  attn_ctx_kernel<<<B_ * H_, 256, 0, stream>>>(ag, Vg, ctxg, N_, D_);

  // 5) attention d (desc attend to nodes, node pad mask on keys)
  attn_scores_kernel<<<B_ * H_, 256, 0, stream>>>(Qd, Kd, pad_mask, sd, D_, N_);
  softmax_kernel<<<B_ * H_ * D_, 256, 0, stream>>>(sd, ad, N_);
  attn_ctx_kernel<<<B_ * H_, 256, 0, stream>>>(ad, Vd, ctxd, D_, N_);

  // 6) outputs
  out_tokens_kernel<<<(int)Mnode, 256, 0, stream>>>(Pn, ctxg, ln_out_g_g, ln_out_g_b, outg);
  out_tokens_kernel<<<(int)Mdesc, 256, 0, stream>>>(Pd, ctxd, ln_out_d_g, ln_out_d_b, outd);
  gvec_kernel<<<B_, 256, 0, stream>>>(outg, pad_mask, gvec);
  dvec_kernel<<<B_, 256, 0, stream>>>(outd, dvec);

  (void)in_sizes; (void)n_in; (void)out_size; (void)ws_size;
}
